// Structure_GraphNetwork_66176856097010
// MI455X (gfx1250) — matrix operand order repaired and checked by reading, not compiled, not run on hardware
//
#include <hip/hip_runtime.h>

// ---------------------------------------------------------------------------
// GraphNetwork on MI455X (gfx1250, wave32, WMMA).
// All GEMMs run on the matrix pipe via v_wmma_f32_16x16x32_f16 (f16 inputs,
// fp32 accumulate). h table (6.4MB) is L2-resident; gathers/scatter-atomics
// stay in L2. Edge kernels are persistent (grid-stride over 128-edge tiles):
// weights staged to LDS once per block, A rows staged per-wave at b128
// granularity, zero padding written once. E % 128 == 0 -> no bounds checks
// in the hot path. edge_attr pre-converted to f16 once (16B/row = 1 chunk).
// ---------------------------------------------------------------------------

#define N_NODES 50000
#define N_EDGES 800000
#define IN_DIM  16
#define HID     64
#define EA_DIM  8
#define NLAYER  3

#define EDGE_TILES (N_EDGES / 128)     // 6250, exact
#define NODE_BLOCKS ((N_NODES + 127) / 128)
#define EDGE_BLOCKS 1024

static_assert(N_EDGES % 128 == 0, "edge kernels assume no tail");

typedef _Float16 h16_t;
typedef __attribute__((ext_vector_type(16))) _Float16 v16h;
typedef __attribute__((ext_vector_type(8)))  _Float16 v8h;
typedef __attribute__((ext_vector_type(4)))  _Float16 v4h;
typedef __attribute__((ext_vector_type(8)))  float    v8f;

#if defined(__HIP_DEVICE_COMPILE__) && \
    !__has_builtin(__builtin_amdgcn_wmma_f32_16x16x32_f16)
#error "missing gfx1250 wmma builtin"
#endif

static __device__ __forceinline__ v8f wmma16(v16h a, v16h b, v8f c) {
  // (neg_a, A, neg_b, B, c_mod, C, reuse_a, reuse_b)
  return __builtin_amdgcn_wmma_f32_16x16x32_f16(false, a, false, b, (short)0, c,
                                                false, false);
}

union U16h { v16h v; v8h h[2]; };

// A fragment (16x32 f16). 16-bit A layout (ISA 7.12.2): lane l holds row
// M=l&15; elements 0..7 = K (8*(l>>4))+0..7, elements 8..15 = K+16.
static __device__ __forceinline__ v16h loadA(const h16_t* base, int strideH) {
  int l = threadIdx.x & 31, m = l & 15, hi = l >> 4;
  const h16_t* p = base + m * strideH + 8 * hi;
  U16h u;
  u.h[0] = *(const v8h*)(p);
  u.h[1] = *(const v8h*)(p + 16);
  return u.v;
}

// B fragment (32x16 f16) from LDS weights stored transposed: sW[n*Kpad + k].
static __device__ __forceinline__ v16h loadB(const h16_t* sW, int Kpad,
                                             int nbase, int kbase) {
  int l = threadIdx.x & 31, n = l & 15, hi = l >> 4;
  const h16_t* p = sW + (nbase + n) * Kpad + kbase + 8 * hi;
  U16h u;
  u.h[0] = *(const v8h*)(p);
  u.h[1] = *(const v8h*)(p + 16);
  return u.v;
}

// Stage global row-major f32 weights W[K][Nc] into LDS f16, transposed and
// zero-padded to [Npad][Kpad].
static __device__ __forceinline__ void stageW(const float* __restrict__ Wg,
                                              h16_t* sW, int K, int Nc,
                                              int Kpad, int Npad) {
  for (int i = threadIdx.x; i < Npad * Kpad; i += blockDim.x) {
    int n = i / Kpad, k = i % Kpad;
    sW[i] = (k < K && n < Nc) ? (h16_t)Wg[k * Nc + n] : (h16_t)0.0f;
  }
}

// Per-wave b128 gather of one 16-edge tile: rows [h_src | h_dst | ea16] as
// 17 16-byte chunks per row (272 chunks / 32 lanes = 9 iterations).
static __device__ __forceinline__ void gatherTile(
    const h16_t* __restrict__ h16, const h16_t* __restrict__ ea16,
    const int* __restrict__ eidx, h16_t* sAw, int* sSrcw, int* sDstw, int e0,
    int l) {
  if (l < 16) sSrcw[l] = eidx[e0 + l];
  else        sDstw[l - 16] = eidx[N_EDGES + e0 + (l - 16)];
  for (int i = l; i < 16 * 17; i += 32) {
    int r = i / 17, ch = i % 17;
    const h16_t* src;
    if (ch < 8)       src = h16 + sSrcw[r] * HID + ch * 8;
    else if (ch < 16) src = h16 + sDstw[r] * HID + (ch - 8) * 8;
    else              src = ea16 + (e0 + r) * EA_DIM;
    *(v8h*)(sAw + r * 160 + ch * 8) = *(const v8h*)src;
  }
}

// ---------------------- edge_attr f32 -> f16 (once) ------------------------
__global__ void cvt_ea_kernel(const float* __restrict__ ea,
                              h16_t* __restrict__ ea16) {
  int i = blockIdx.x * blockDim.x + threadIdx.x; // one float4 per thread
  if (i < N_EDGES * EA_DIM / 4) {
    float4 g = ((const float4*)ea)[i];
    v4h h;
    h[0] = (h16_t)g.x; h[1] = (h16_t)g.y; h[2] = (h16_t)g.z; h[3] = (h16_t)g.w;
    ((v4h*)ea16)[i] = h;
  }
}

// ------------------------------- encoder -----------------------------------
__global__ void enc_kernel(const float* __restrict__ x,
                           const float* __restrict__ Wenc,
                           const float* __restrict__ benc,
                           h16_t* __restrict__ h16) {
  __shared__ __align__(16) h16_t sW[64 * 32];
  __shared__ __align__(16) h16_t sA[8][16 * 32];
  int tid = threadIdx.x, w = tid >> 5, l = tid & 31;
  stageW(Wenc, sW, IN_DIM, HID, 32, 64);
  int row0 = blockIdx.x * 128 + w * 16;
  bool full = (row0 + 16 <= N_NODES);  // wave-uniform fast path
  for (int i = l; i < 16 * 32; i += 32) {
    int r = i >> 5, c = i & 31;
    int node = row0 + r;
    sA[w][i] = (c < IN_DIM && (full || node < N_NODES))
                   ? (h16_t)x[node * IN_DIM + c]
                   : (h16_t)0.0f;
  }
  __syncthreads();
  v8f zf = {};
  v8f acc[4] = {zf, zf, zf, zf};
  v16h a = loadA(&sA[w][0], 32);
#pragma unroll
  for (int ct = 0; ct < 4; ++ct) {
    v16h b = loadB(sW, 32, ct * 16, 0);
    acc[ct] = wmma16(a, b, acc[ct]);
  }
  int n = l & 15, hi = l >> 4;
#pragma unroll
  for (int ct = 0; ct < 4; ++ct) {
    int col = ct * 16 + n;
    float bias = benc[col];
    if (full) {
#pragma unroll
      for (int r = 0; r < 8; ++r)
        h16[(row0 + r + 8 * hi) * HID + col] = (h16_t)(acc[ct][r] + bias);
    } else {
#pragma unroll
      for (int r = 0; r < 8; ++r) {
        int node = row0 + r + 8 * hi;
        if (node < N_NODES) h16[node * HID + col] = (h16_t)(acc[ct][r] + bias);
      }
    }
  }
}

// --------------------- message + segment-sum (per layer) -------------------
__global__ void msg_kernel(const h16_t* __restrict__ h16,
                           const int* __restrict__ eidx,
                           const h16_t* __restrict__ ea16,
                           const float* __restrict__ Wmsg,
                           const float* __restrict__ bmsg,
                           float* __restrict__ agg) {
  __shared__ __align__(16) h16_t sW[64 * 160];    // 20.0 KB
  __shared__ __align__(16) h16_t sA[8][16 * 160]; // 40.0 KB (per-wave slices)
  __shared__ int sSrc[8][16], sDst[8][16];
  int tid = threadIdx.x, w = tid >> 5, l = tid & 31;
  int n = l & 15, hi = l >> 4;
  h16_t* sAw = &sA[w][0];
  stageW(Wmsg, sW, 2 * HID + EA_DIM, HID, 160, 64);
  // zero the constant pad columns [136,160) once
  for (int i = l; i < 16 * 12; i += 32) {
    int r = i / 12, q = i % 12;
    *(uint32_t*)(sAw + r * 160 + 136 + q * 2) = 0;
  }
  __syncthreads();
  float biasv[4];
#pragma unroll
  for (int ct = 0; ct < 4; ++ct) biasv[ct] = bmsg[ct * 16 + n];

  for (int tile = blockIdx.x; tile < EDGE_TILES; tile += gridDim.x) {
    int e0 = tile * 128 + w * 16;
    gatherTile(h16, ea16, eidx, sAw, sSrc[w], sDst[w], e0, l);
    v8f zf = {};
    v8f acc[4] = {zf, zf, zf, zf};
#pragma unroll
    for (int kt = 0; kt < 5; ++kt) {
      v16h a = loadA(sAw + kt * 32, 160);
#pragma unroll
      for (int ct = 0; ct < 4; ++ct) {
        v16h b = loadB(sW, 160, ct * 16, kt * 32);
        acc[ct] = wmma16(a, b, acc[ct]);
      }
    }
#pragma unroll
    for (int ct = 0; ct < 4; ++ct) {
      int col = ct * 16 + n;
#pragma unroll
      for (int r = 0; r < 8; ++r) {
        float v = fmaxf(acc[ct][r] + biasv[ct], 0.0f);
        atomicAdd(&agg[sDst[w][r + 8 * hi] * HID + col], v); // L2 scatter-add
      }
    }
  }
}

// ----------------------------- node update ---------------------------------
__global__ void upd_kernel(h16_t* __restrict__ h16,
                           const float* __restrict__ agg,
                           const float* __restrict__ Wupd,
                           const float* __restrict__ bupd) {
  __shared__ __align__(16) h16_t sW[64 * 128];    // 16 KB
  __shared__ __align__(16) h16_t sA[8][16 * 128]; // 32 KB
  int tid = threadIdx.x, w = tid >> 5, l = tid & 31;
  h16_t* sAw = &sA[w][0];
  stageW(Wupd, sW, 2 * HID, HID, 128, 64);
  int n0 = blockIdx.x * 128 + w * 16;
  bool full = (n0 + 16 <= N_NODES);
  // h half: 16 rows x 8 b128 chunks
  v8h z8 = {};
  for (int i = l; i < 16 * 8; i += 32) {
    int r = i >> 3, ch = i & 7;
    int node = n0 + r;
    v8h v = (full || node < N_NODES)
                ? *(const v8h*)(h16 + node * HID + ch * 8)
                : z8;
    *(v8h*)(sAw + r * 128 + ch * 8) = v;
  }
  // agg half: 16 rows x 16 float4 chunks -> v4h
  for (int i = l; i < 16 * 16; i += 32) {
    int r = i >> 4, q = i & 15;
    int node = n0 + r;
    v4h h = {};
    if (full || node < N_NODES) {
      float4 g = *(const float4*)(agg + node * HID + q * 4);
      h[0] = (h16_t)g.x; h[1] = (h16_t)g.y;
      h[2] = (h16_t)g.z; h[3] = (h16_t)g.w;
    }
    *(v4h*)(sAw + r * 128 + 64 + q * 4) = h;
  }
  __syncthreads();
  v8f zf = {};
  v8f acc[4] = {zf, zf, zf, zf};
#pragma unroll
  for (int kt = 0; kt < 4; ++kt) {
    v16h a = loadA(sAw + kt * 32, 128);
#pragma unroll
    for (int ct = 0; ct < 4; ++ct) {
      v16h b = loadB(sW, 128, ct * 16, kt * 32);
      acc[ct] = wmma16(a, b, acc[ct]);
    }
  }
  int n = l & 15, hi = l >> 4;
#pragma unroll
  for (int ct = 0; ct < 4; ++ct) {
    int col = ct * 16 + n;
    float bias = bupd[col];
    if (full) {
#pragma unroll
      for (int r = 0; r < 8; ++r)
        h16[(n0 + r + 8 * hi) * HID + col] =
            (h16_t)fmaxf(acc[ct][r] + bias, 0.0f);
    } else {
#pragma unroll
      for (int r = 0; r < 8; ++r) {
        int node = n0 + r + 8 * hi;
        if (node < N_NODES)
          h16[node * HID + col] = (h16_t)fmaxf(acc[ct][r] + bias, 0.0f);
      }
    }
  }
}

// ------------------------------ node head ----------------------------------
__global__ void node_head_kernel(const h16_t* __restrict__ h16,
                                 const float* __restrict__ Wnd1,
                                 const float* __restrict__ bnd1,
                                 const float* __restrict__ Wnd2,
                                 const float* __restrict__ bnd2,
                                 float* __restrict__ nout) {
  __shared__ __align__(16) h16_t sW1[64 * 64];
  __shared__ __align__(16) h16_t sW2[16 * 64];
  __shared__ __align__(16) h16_t sT[8][16 * 64];
  int tid = threadIdx.x, w = tid >> 5, l = tid & 31;
  stageW(Wnd1, sW1, 64, 64, 64, 64);
  stageW(Wnd2, sW2, 64, 2, 64, 16);
  int n0 = blockIdx.x * 128 + w * 16;
  bool full = (n0 + 16 <= N_NODES);
  __syncthreads();
  int n = l & 15, hi = l >> 4;
  v16h az;
#pragma unroll
  for (int i = 0; i < 16; ++i) az[i] = (_Float16)0;
  v8f zf = {};
  v8f acc[4] = {zf, zf, zf, zf};
#pragma unroll
  for (int kt = 0; kt < 2; ++kt) {
    // A fragments straight from global h16 (rows are contiguous)
    U16h u;
    int node = n0 + n; // M = lane&15
    if (full || node < N_NODES) {
      const h16_t* p = h16 + node * HID + kt * 32 + 8 * hi;
      u.h[0] = *(const v8h*)p;
      u.h[1] = *(const v8h*)(p + 16);
    } else {
      u.v = az;
    }
#pragma unroll
    for (int ct = 0; ct < 4; ++ct) {
      v16h b = loadB(sW1, 64, ct * 16, kt * 32);
      acc[ct] = wmma16(u.v, b, acc[ct]);
    }
  }
  // t1 = relu(h@W1+b1) -> LDS, re-read as A fragments for the 64->2 GEMM
#pragma unroll
  for (int ct = 0; ct < 4; ++ct) {
    float bias = bnd1[ct * 16 + n];
#pragma unroll
    for (int r = 0; r < 8; ++r)
      sT[w][(r + 8 * hi) * 64 + ct * 16 + n] =
          (h16_t)fmaxf(acc[ct][r] + bias, 0.0f);
  }
  v8f acc2 = zf;
#pragma unroll
  for (int kt = 0; kt < 2; ++kt) {
    v16h a2 = loadA(&sT[w][kt * 32], 64);
    v16h b2 = loadB(sW2, 64, 0, kt * 32);
    acc2 = wmma16(a2, b2, acc2);
  }
  if (n < 2) {
    float bias = bnd2[n];
    if (full) {
#pragma unroll
      for (int r = 0; r < 8; ++r)
        nout[(n0 + r + 8 * hi) * 2 + n] = acc2[r] + bias;
    } else {
#pragma unroll
      for (int r = 0; r < 8; ++r) {
        int node = n0 + r + 8 * hi;
        if (node < N_NODES) nout[node * 2 + n] = acc2[r] + bias;
      }
    }
  }
}

// ------------------------------ edge head ----------------------------------
__global__ void edge_head_kernel(const h16_t* __restrict__ h16,
                                 const int* __restrict__ eidx,
                                 const h16_t* __restrict__ ea16,
                                 const float* __restrict__ Wed1,
                                 const float* __restrict__ bed1,
                                 const float* __restrict__ Wed2,
                                 const float* __restrict__ bed2,
                                 const float* __restrict__ Wed3,
                                 const float* __restrict__ bed3,
                                 float* __restrict__ eout) {
  __shared__ __align__(16) h16_t sW1[128 * 160]; // 40 KB
  __shared__ __align__(16) h16_t sW2[64 * 128];  // 16 KB
  __shared__ __align__(16) h16_t sW3[16 * 64];   //  2 KB
  __shared__ __align__(16) h16_t sA[8][16 * 160];// 40 KB (reused for e1/e2)
  __shared__ int sSrc[8][16], sDst[8][16];
  int tid = threadIdx.x, w = tid >> 5, l = tid & 31;
  int n = l & 15, hi = l >> 4;
  h16_t* sAw = &sA[w][0];
  stageW(Wed1, sW1, 2 * HID + EA_DIM, 128, 160, 128);
  stageW(Wed2, sW2, 128, 64, 128, 64);
  stageW(Wed3, sW3, 64, 6, 64, 16);
  for (int i = l; i < 16 * 12; i += 32) { // constant zero pad, once
    int r = i / 12, q = i % 12;
    *(uint32_t*)(sAw + r * 160 + 136 + q * 2) = 0;
  }
  __syncthreads();
  float b1v[8], b2v[4];
#pragma unroll
  for (int ct = 0; ct < 8; ++ct) b1v[ct] = bed1[ct * 16 + n];
#pragma unroll
  for (int ct = 0; ct < 4; ++ct) b2v[ct] = bed2[ct * 16 + n];
  float b3 = bed3[n < 6 ? n : 0];

  for (int tile = blockIdx.x; tile < EDGE_TILES; tile += gridDim.x) {
    int e0 = tile * 128 + w * 16;
    gatherTile(h16, ea16, eidx, sAw, sSrc[w], sDst[w], e0, l);
    v8f zf = {};
    // e1 = relu(ef @ W_ed1 + b1): 5 K-tiles x 8 N-tiles
    v8f a8[8] = {zf, zf, zf, zf, zf, zf, zf, zf};
#pragma unroll
    for (int kt = 0; kt < 5; ++kt) {
      v16h a = loadA(sAw + kt * 32, 160);
#pragma unroll
      for (int ct = 0; ct < 8; ++ct) {
        v16h b = loadB(sW1, 160, ct * 16, kt * 32);
        a8[ct] = wmma16(a, b, a8[ct]);
      }
    }
    // e1 back into this wave's LDS slice (stride 128); but careful: the
    // stride-128 region overlaps the stride-160 A rows INCLUDING the pad
    // columns of rows 0..11 -- pad lives at r*160+[136,160) which for r<12
    // lies inside [0, 2048). Writes below cover [0,2048) fully each tile,
    // and the next gatherTile rewrites [r*160, r*160+136) fully, so the only
    // cells that must stay zero are the pads -- they are NOT in [0,2048)
    // for r>=13? r=12: 12*160+136=2056 >= 2048. r<12: pad at r*160+136 <
    // 2048 gets clobbered by e1 -- so re-zero pads for rows 0..11 after e2
    // phase below (12*12=144 u32, 5 iters) before next tile's gather? No:
    // gatherTile rewrites only [r*160, r*160+136); pads must persist. We
    // re-zero them at the end of each tile iteration (cheap).
    // e1 values:
#pragma unroll
    for (int ct = 0; ct < 8; ++ct) {
#pragma unroll
      for (int r = 0; r < 8; ++r)
        sAw[(r + 8 * hi) * 128 + ct * 16 + n] =
            (h16_t)fmaxf(a8[ct][r] + b1v[ct], 0.0f);
    }
    // e2 = relu(e1 @ W_ed2 + b2): 4 x 4
    v8f a4[4] = {zf, zf, zf, zf};
#pragma unroll
    for (int kt = 0; kt < 4; ++kt) {
      v16h a = loadA(sAw + kt * 32, 128);
#pragma unroll
      for (int ct = 0; ct < 4; ++ct) {
        v16h b = loadB(sW2, 128, ct * 16, kt * 32);
        a4[ct] = wmma16(a, b, a4[ct]);
      }
    }
#pragma unroll
    for (int ct = 0; ct < 4; ++ct) {
#pragma unroll
      for (int r = 0; r < 8; ++r)
        sAw[(r + 8 * hi) * 64 + ct * 16 + n] =
            (h16_t)fmaxf(a4[ct][r] + b2v[ct], 0.0f);
    }
    // e3 = e2 @ W_ed3 + b3 (6 valid cols, padded to 16)
    v8f a1 = zf;
#pragma unroll
    for (int kt = 0; kt < 2; ++kt) {
      v16h a = loadA(sAw + kt * 32, 64);
      v16h b = loadB(sW3, 64, 0, kt * 32);
      a1 = wmma16(a, b, a1);
    }
    if (n < 6) {
#pragma unroll
      for (int r = 0; r < 8; ++r)
        eout[(e0 + r + 8 * hi) * 6 + n] = a1[r] + b3;
    }
    // restore the constant zero pads clobbered by the e1/e2 LDS reuse
    for (int i = l; i < 16 * 12; i += 32) {
      int r = i / 12, q = i % 12;
      *(uint32_t*)(sAw + r * 160 + 136 + q * 2) = 0;
    }
  }
}

// ------------------------------- utility -----------------------------------
__global__ void zero_kernel(float4* __restrict__ p, int n4) {
  int i = blockIdx.x * blockDim.x + threadIdx.x;
  if (i < n4) p[i] = make_float4(0.0f, 0.0f, 0.0f, 0.0f);
}

// ---------------------------------------------------------------------------
extern "C" void kernel_launch(void* const* d_in, const int* in_sizes, int n_in,
                              void* d_out, int out_size, void* d_ws,
                              size_t ws_size, hipStream_t stream) {
  const float* x    = (const float*)d_in[0];
  const int*   eidx = (const int*)d_in[1]; // JAX default x64-off -> int32
  const float* ea   = (const float*)d_in[2];
  const float* Wenc = (const float*)d_in[3];
  const float* benc = (const float*)d_in[4];
  const float* Wmsg = (const float*)d_in[5];
  const float* bmsg = (const float*)d_in[6];
  const float* Wupd = (const float*)d_in[7];
  const float* bupd = (const float*)d_in[8];
  const float* Wnd1 = (const float*)d_in[9];
  const float* bnd1 = (const float*)d_in[10];
  const float* Wnd2 = (const float*)d_in[11];
  const float* bnd2 = (const float*)d_in[12];
  const float* Wed1 = (const float*)d_in[13];
  const float* bed1 = (const float*)d_in[14];
  const float* Wed2 = (const float*)d_in[15];
  const float* bed2 = (const float*)d_in[16];
  const float* Wed3 = (const float*)d_in[17];
  const float* bed3 = (const float*)d_in[18];

  // workspace: h_f16 [N,64] | agg f32 [N,64] | ea16 f16 [E,8]
  char* ws = (char*)d_ws;
  h16_t* h16 = (h16_t*)ws;
  float* agg = (float*)(ws + (size_t)N_NODES * HID * sizeof(h16_t));
  h16_t* ea16 = (h16_t*)(ws + (size_t)N_NODES * HID * sizeof(h16_t) +
                         (size_t)N_NODES * HID * sizeof(float));

  float* nout = (float*)d_out;              // [N,2]
  float* eout = nout + 2 * (size_t)N_NODES; // [E,6]

  dim3 blk(256);
  int aggN4 = N_NODES * HID / 4;      // 800000
  int eaN4 = N_EDGES * EA_DIM / 4;    // 1600000

  cvt_ea_kernel<<<(eaN4 + 255) / 256, blk, 0, stream>>>(ea, ea16);
  enc_kernel<<<NODE_BLOCKS, blk, 0, stream>>>(x, Wenc, benc, h16);
  for (int ly = 0; ly < NLAYER; ++ly) {
    zero_kernel<<<(aggN4 + 255) / 256, blk, 0, stream>>>((float4*)agg, aggN4);
    msg_kernel<<<EDGE_BLOCKS, blk, 0, stream>>>(h16, eidx, ea16, Wmsg, bmsg,
                                                agg);
    upd_kernel<<<NODE_BLOCKS, blk, 0, stream>>>(h16, agg, Wupd, bupd);
  }
  node_head_kernel<<<NODE_BLOCKS, blk, 0, stream>>>(h16, Wnd1, bnd1, Wnd2,
                                                    bnd2, nout);
  edge_head_kernel<<<EDGE_BLOCKS, blk, 0, stream>>>(h16, eidx, ea16, Wed1,
                                                    bed1, Wed2, bed2, Wed3,
                                                    bed3, eout);
}
